// GraphSAGE_46050639348070
// MI455X (gfx1250) — matrix operand order repaired and checked by reading, not compiled, not run on hardware
//
#include <hip/hip_runtime.h>
#include <hip/hip_bf16.h>
#include <stdint.h>
#include <stddef.h>

// GraphSAGE 2-layer forward for MI455X (gfx1250, wave32, WMMA bf16 path).
// N=50000 nodes, E=800000 edges, D_IN=128, D_H=256, D_OUT=128.

#define NN 50000
#define EE 800000

typedef __attribute__((ext_vector_type(16))) __bf16 v16bf;
typedef __attribute__((ext_vector_type(8)))  __bf16 v8bf;
typedef __attribute__((ext_vector_type(8)))  float  v8f;

// ---------------------------------------------------------------- zero
__global__ void zero_f32(float* __restrict__ p, int n) {
  int i = blockIdx.x * blockDim.x + threadIdx.x;
  int stride = gridDim.x * blockDim.x;
  for (; i < n; i += stride) p[i] = 0.0f;
}

// ---------------------------------------------------- pack [Wl|Wr] -> bf16
// Wcat[r][k] = (k < K) ? Wl[r][k] : Wr[r][k-K], rows x 2K, bf16 row-major.
__global__ void prep_weights(const float* __restrict__ Wl,
                             const float* __restrict__ Wr,
                             __bf16* __restrict__ Wcat, int rows, int K) {
  int i = blockIdx.x * blockDim.x + threadIdx.x;
  int total = rows * 2 * K;
  if (i >= total) return;
  int r = i / (2 * K);
  int k = i - r * (2 * K);
  float v = (k < K) ? Wl[(size_t)r * K + k] : Wr[(size_t)r * K + (k - K)];
  Wcat[i] = (__bf16)v;
}

// ---------------------------------------------------------------- 1/deg
__global__ void recip_deg(float* __restrict__ deg, int n) {
  int i = blockIdx.x * blockDim.x + threadIdx.x;
  if (i < n) deg[i] = 1.0f / fmaxf(deg[i], 1.0f);
}

// ------------------------------------------------------------- scatter-add
// One wave per edge; each lane handles 4 consecutive floats per 128-chunk.
// f32 atomics resolve in the L2 atomic units (agg buffers are L2-resident).
template<int D, bool COUNT>
__global__ void scatter_add(const float* __restrict__ feat,
                            const int*   __restrict__ ei,   // [2][E]
                            float* __restrict__ agg,
                            float* __restrict__ deg) {
  int t = blockIdx.x * blockDim.x + threadIdx.x;
  int e = t >> 5;
  int lane = t & 31;
  if (e >= EE) return;
  int src = ei[e];
  int dst = ei[EE + e];
  const float* xr = feat + (size_t)src * D;
  float*       ar = agg  + (size_t)dst * D;
#pragma unroll
  for (int j = 0; j < D / 128; ++j) {
    int o = j * 128 + lane * 4;
    float4 v = *(const float4*)(xr + o);
    __hip_atomic_fetch_add(ar + o + 0, v.x, __ATOMIC_RELAXED, __HIP_MEMORY_SCOPE_AGENT);
    __hip_atomic_fetch_add(ar + o + 1, v.y, __ATOMIC_RELAXED, __HIP_MEMORY_SCOPE_AGENT);
    __hip_atomic_fetch_add(ar + o + 2, v.z, __ATOMIC_RELAXED, __HIP_MEMORY_SCOPE_AGENT);
    __hip_atomic_fetch_add(ar + o + 3, v.w, __ATOMIC_RELAXED, __HIP_MEMORY_SCOPE_AGENT);
  }
  if (COUNT && lane == 0)
    __hip_atomic_fetch_add(deg + dst, 1.0f, __ATOMIC_RELAXED, __HIP_MEMORY_SCOPE_AGENT);
}

// ----------------------------------------------------------------- GEMM
// out[m][n] = act( [mean(m)|x(m)] . Wcat[n][:] + bias[n] )
// Block = 256 threads = 8 waves; one block owns a 16-row panel and
// 8*16*NW output columns (layer1: NW=2 -> 256 cols; layer2: NW=1 -> 128).
// The 16 x KK bf16 A-panel (mean half pre-scaled by 1/deg) is built ONCE
// per block in LDS, then each wave reads its fragments via ds loads and
// reuses one A fragment for NW WMMAs.
// A frag k order used here (contiguous, not ISA table order): lane<16 -> K k0..k0+15,
// lane>=16 -> +16.  B frag layout: lane<16 -> K k0..k0+15, lane>=16 -> +16.
template<int K_IN, int NW, bool RELU>
__global__ __launch_bounds__(256)
void sage_gemm(const float*  __restrict__ agg,
               const float*  __restrict__ rdeg,
               const float*  __restrict__ xin,
               const __bf16* __restrict__ Wcat,   // [N_OUT][2*K_IN] bf16
               const float*  __restrict__ bias,
               float*        __restrict__ out) {
  constexpr int KK   = 2 * K_IN;
  constexpr int LDA  = KK + 8;          // pad 8 bf16 (16B) -> no bank clash
  constexpr int NOUT = 8 * 16 * NW;     // 256 (layer1) or 128 (layer2)
  __shared__ __bf16 As[16 * LDA];

  const int tid   = threadIdx.x;
  const int mbase = blockIdx.x * 16;

  // ---- cooperative A-panel fill: 16 rows x KK bf16, pairs per thread ----
  // pair index p -> row r = p / (KK/2), k = 2*(p % (KK/2)); trip count is a
  // multiple of blockDim so EXEC stays all-ones (WMMA requirement).
  for (int p = tid; p < 16 * (KK / 2); p += 256) {
    int r  = p / (KK / 2);
    int k  = (p - r * (KK / 2)) * 2;
    int row = mbase + r;
    float2 v;
    float  sc;
    if (k < K_IN) {
      v  = *(const float2*)(agg + (size_t)row * K_IN + k);
      sc = rdeg[row];
    } else {
      v  = *(const float2*)(xin + (size_t)row * K_IN + (k - K_IN));
      sc = 1.0f;
    }
    __bf16 lo = (__bf16)(v.x * sc);
    __bf16 hi = (__bf16)(v.y * sc);
    // packed 4B store (k is even)
    unsigned int pk = ((unsigned int)*(unsigned short*)&hi << 16) |
                      (unsigned int)*(unsigned short*)&lo;
    *(unsigned int*)(&As[r * LDA + k]) = pk;
  }
  __syncthreads();

  const int wave = tid >> 5;
  const int lane = tid & 31;
  const int l16  = lane & 15;
  const int hi   = lane >> 4;           // half-wave select

  // accumulators pre-loaded with bias (each lane owns one output column)
  v8f c[NW];
  int ncol[NW];
#pragma unroll
  for (int j = 0; j < NW; ++j) {
    ncol[j] = wave * 16 * NW + j * 16 + l16;
    float bn = bias[ncol[j]];
#pragma unroll
    for (int v = 0; v < 8; ++v) c[j][v] = bn;
  }

  const __bf16* arow = &As[l16 * LDA];
  for (int k0 = 0; k0 < KK; k0 += 32) {
    // A fragment: two 16B LDS reads per lane
    v8bf alo = *(const v8bf*)(arow + k0 + hi * 16);
    v8bf ahi = *(const v8bf*)(arow + k0 + hi * 16 + 8);
    v16bf a;
#pragma unroll
    for (int q = 0; q < 8; ++q) { a[q] = alo[q]; a[8 + q] = ahi[q]; }
#pragma unroll
    for (int j = 0; j < NW; ++j) {
      // B fragment: 16 contiguous bf16 (32B) from weight row = B column
      v16bf b = *(const v16bf*)(Wcat + (size_t)ncol[j] * KK + k0 + hi * 16);
      c[j] = __builtin_amdgcn_wmma_f32_16x16x32_bf16(
          false, a, false, b, (short)0, c[j], false, false);
    }
  }

  // D layout: lane l, vgpr v -> row = v + (l<16?0:8), col = l&15
#pragma unroll
  for (int j = 0; j < NW; ++j) {
#pragma unroll
    for (int v = 0; v < 8; ++v) {
      float val = c[j][v];
      if (RELU) val = fmaxf(val, 0.0f);
      int m = mbase + v + hi * 8;
      out[(size_t)m * NOUT + ncol[j]] = val;
    }
  }
}

// ----------------------------------------------------------------- launch
extern "C" void kernel_launch(void* const* d_in, const int* in_sizes, int n_in,
                              void* d_out, int out_size, void* d_ws, size_t ws_size,
                              hipStream_t stream) {
  const float* x   = (const float*)d_in[0];   // [N,128]
  const int*   ei  = (const int*)  d_in[1];   // [2,E]
  const float* W1l = (const float*)d_in[2];   // [256,128]
  const float* b1l = (const float*)d_in[3];   // [256]
  const float* W1r = (const float*)d_in[4];   // [256,128]
  const float* W2l = (const float*)d_in[5];   // [128,256]
  const float* b2l = (const float*)d_in[6];   // [128]
  const float* W2r = (const float*)d_in[7];   // [128,256]
  float* out = (float*)d_out;                 // [N,128]

  // workspace layout (all 32B-aligned)
  float*  agg1 = (float*)d_ws;                         // N*128
  float*  agg2 = agg1 + (size_t)NN * 128;              // N*256
  float*  deg  = agg2 + (size_t)NN * 256;              // N (becomes rdeg)
  float*  h    = deg  + ((NN + 63) / 64) * 64;         // N*256
  __bf16* Wc1  = (__bf16*)(h + (size_t)NN * 256);      // 256 x 256
  __bf16* Wc2  = Wc1 + 256 * 256;                      // 128 x 512

  // 1) zero agg1|agg2|deg (contiguous N*385 floats)
  zero_f32<<<4096, 256, 0, stream>>>(agg1, NN * 385);

  // 2) bf16 weight packing
  prep_weights<<<(256 * 256 + 255) / 256, 256, 0, stream>>>(W1l, W1r, Wc1, 256, 128);
  prep_weights<<<(128 * 512 + 255) / 256, 256, 0, stream>>>(W2l, W2r, Wc2, 128, 256);

  // 3) layer-1 mean aggregation (one wave per edge) + degree
  scatter_add<128, true><<<(EE * 32) / 256, 256, 0, stream>>>(x, ei, agg1, deg);
  recip_deg<<<(NN + 255) / 256, 256, 0, stream>>>(deg, NN);

  // 4) layer-1 GEMM + bias + ReLU -> h [N,256]  (8 waves x 32 cols)
  sage_gemm<128, 2, true><<<dim3(NN / 16, 1), 256, 0, stream>>>(
      agg1, deg, x, Wc1, b1l, h);

  // 5) layer-2 aggregation on h (reuse rdeg)
  scatter_add<256, false><<<(EE * 32) / 256, 256, 0, stream>>>(h, ei, agg2, deg);

  // 6) layer-2 GEMM + bias -> out [N,128]  (8 waves x 16 cols)
  sage_gemm<256, 1, false><<<dim3(NN / 16, 1), 256, 0, stream>>>(
      agg2, deg, h, Wc2, b2l, out);
}
